// Seq2SeqRNN_62491774156956
// MI455X (gfx1250) — matrix operand order repaired and checked by reading, not compiled, not run on hardware
//
#include <hip/hip_runtime.h>
#include <hip/hip_bf16.h>
#include <math.h>

typedef __bf16 bf16_t;
typedef __attribute__((ext_vector_type(16))) __bf16 v16bf;
typedef __attribute__((ext_vector_type(8)))  float  v8f;

#define B_SZ  64
#define S_LEN 128
#define T_LEN 64
#define E_DIM 256
#define H_DIM 512
#define V_DIM 256

// ---------------------------------------------------------------------------
// Convert + concatenate two fp32 weight blocks (N x K0, N x K1) into one
// bf16 (N x (K0+K1)) matrix. Run once per launch; stays resident in L2.
// ---------------------------------------------------------------------------
__global__ void k_cvt_cat(const float* __restrict__ s0, int K0,
                          const float* __restrict__ s1, int K1,
                          bf16_t* __restrict__ dst, int N) {
  int K = K0 + K1;
  int gid = blockIdx.x * blockDim.x + threadIdx.x;
  int total = N * K;
  if (gid >= total) return;
  int o = gid / K, k = gid - o * K;
  float v = (k < K0) ? s0[(size_t)o * K0 + k] : s1[(size_t)o * K1 + (k - K0)];
  dst[gid] = (bf16_t)v;
}

__global__ void k_zero(float* __restrict__ p, int n) {
  int i = blockIdx.x * blockDim.x + threadIdx.x;
  if (i < n) p[i] = 0.0f;
}

// d1[b][h] = enc_out[b][S-1][h]  (fp32 + bf16 mirror)
__global__ void k_copy_last(const float* __restrict__ enc_out,
                            float* __restrict__ d1, bf16_t* __restrict__ d1_bf) {
  int i = blockIdx.x * blockDim.x + threadIdx.x;   // 64*512
  if (i >= B_SZ * H_DIM) return;
  int b = i >> 9, h = i & (H_DIM - 1);
  float v = enc_out[((size_t)b * S_LEN + (S_LEN - 1)) * H_DIM + h];
  d1[i] = v;
  d1_bf[i] = (bf16_t)v;
}

// ---------------------------------------------------------------------------
// Build fused GEMM input A (B_SZ x Ktot, bf16) =
//   [ emb[idx[b][t]] (E) | s1 (K1) | s2 (K2) ]
// s2_valid==0 -> zeros for the s2 segment (initial hidden state).
// Writing bf16 here removes all fp32->bf16 conversion from the GEMM K-loop.
// ---------------------------------------------------------------------------
__global__ void k_build_A(const float* __restrict__ emb, const int* __restrict__ idx,
                          int idx_stride, int t, int E,
                          const float* __restrict__ s1, int ld1, int K1,
                          const float* __restrict__ s2, int ld2, int K2, int s2_valid,
                          bf16_t* __restrict__ A, int Ktot) {
  int gid = blockIdx.x * blockDim.x + threadIdx.x;
  int total = B_SZ * Ktot;
  if (gid >= total) return;
  int b = gid / Ktot, k = gid - b * Ktot;
  float v;
  if (k < E) {
    int tok = idx[b * idx_stride + t];
    v = emb[(size_t)tok * E + k];
  } else if (k < E + K1) {
    v = s1[(size_t)b * ld1 + (k - E)];
  } else {
    v = s2_valid ? s2[(size_t)b * ld2 + (k - E - K1)] : 0.0f;
  }
  A[gid] = (bf16_t)v;
}

// ---------------------------------------------------------------------------
// WMMA GEMM: out[m][n] = act( A[m][:] . W[n][:] + bias[n] )   (i.e. A @ W^T)
//   A : bf16 (M x lda row-major)   W : bf16 (N x K row-major)
//   out: fp32 with row stride ldo; optional bf16 mirror out_bf (row stride
//   ldo_bf) so downstream GEMMs can consume the result without conversion.
// One wave computes one 16x16 tile with v_wmma_f32_16x16x32_bf16.
// Block = 128 threads = 4 waves = 4 consecutive M tiles; grid = (N/16, M/64).
// ---------------------------------------------------------------------------
__global__ void k_gemm_bf16(const bf16_t* __restrict__ A, int lda,
                            const bf16_t* __restrict__ W, int K,
                            const float* __restrict__ bias,
                            float* __restrict__ out, long ldo,
                            bf16_t* __restrict__ out_bf, long ldo_bf,
                            int apply_tanh) {
  int lane = threadIdx.x & 31;
  int wave = threadIdx.x >> 5;
  int l15  = lane & 15;
  int half = lane >> 4;
  int mBase = (blockIdx.y * 4 + wave) * 16;
  int nBase = blockIdx.x * 16;

  const bf16_t* Arow = A + (size_t)(mBase + l15) * lda;
  const bf16_t* Wrow = W + (size_t)(nBase + l15) * K;

  v8f acc = {};
  for (int k0 = 0; k0 < K; k0 += 32) {
    v16bf a, b;
    // A 16x32 bf16 layout: lanes 0-15: M=lane, elems 0-15 -> K k0+0..15;
    // lanes 16-31: same M, K k0+16..31 (matches B).
    // 16 contiguous bf16 -> two 16B loads.
#pragma unroll
    for (int i = 0; i < 16; ++i) a[i] = Arow[k0 + half * 16 + i];
    // B 32x16 layout: lanes 0-15 hold K k0+0..15 (N = lane),
    // lanes 16-31 hold K k0+16..31. 16 contiguous bf16 -> two 16B loads.
#pragma unroll
    for (int i = 0; i < 16; ++i) b[i] = Wrow[k0 + half * 16 + i];
    acc = __builtin_amdgcn_wmma_f32_16x16x32_bf16(
        /*neg_a=*/false, a, /*neg_b=*/false, b,
        /*c_mod=*/(short)0, acc, /*reuse_a=*/false, /*reuse_b=*/false);
  }

  int n = nBase + l15;
  float bv = bias ? bias[n] : 0.0f;
#pragma unroll
  for (int r = 0; r < 8; ++r) {
    int m = mBase + half * 8 + r;      // C/D layout: VGPR r -> M = r (+8 for hi lanes)
    float v = acc[r] + bv;
    if (apply_tanh) v = tanhf(v);
    out[(size_t)m * ldo + n] = v;
    if (out_bf) out_bf[(size_t)m * ldo_bf + n] = (bf16_t)v;
  }
}

// ---------------------------------------------------------------------------
// Attention for one decoder step. One block per batch row b (grid=64, 256 thr):
//   score[s] = v_a . tanh(q[b] + U_enc[b][s]);  w = softmax(score)
//   ctx[b]   = sum_s w[s] * enc_out[b][s]
// ---------------------------------------------------------------------------
__global__ void k_attention(const float* __restrict__ q,
                            const float* __restrict__ Uenc,
                            const float* __restrict__ enc,
                            const float* __restrict__ va,
                            float* __restrict__ ctx) {
  __shared__ float sc[S_LEN];
  __shared__ float redMax, redInv;
  int b = blockIdx.x;
  int tid = threadIdx.x;
  int lane = tid & 31, wv = tid >> 5;          // 8 waves
  const float* qb = q + (size_t)b * H_DIM;

  for (int s = wv; s < S_LEN; s += 8) {
    const float* Ub = Uenc + ((size_t)b * S_LEN + s) * H_DIM;
    float sum = 0.0f;
    for (int h = lane; h < H_DIM; h += 32)
      sum += tanhf(qb[h] + Ub[h]) * va[h];
    for (int off = 16; off; off >>= 1) sum += __shfl_xor(sum, off, 32);
    if (lane == 0) sc[s] = sum;
  }
  __syncthreads();

  if (tid < 32) {
    float m = -1e30f;
    for (int s = lane; s < S_LEN; s += 32) m = fmaxf(m, sc[s]);
    for (int off = 16; off; off >>= 1) m = fmaxf(m, __shfl_xor(m, off, 32));
    if (lane == 0) redMax = m;
  }
  __syncthreads();
  float mx = redMax;
  if (tid < S_LEN) sc[tid] = expf(sc[tid] - mx);
  __syncthreads();
  if (tid < 32) {
    float s = 0.0f;
    for (int i = lane; i < S_LEN; i += 32) s += sc[i];
    for (int off = 16; off; off >>= 1) s += __shfl_xor(s, off, 32);
    if (lane == 0) redInv = 1.0f / s;
  }
  __syncthreads();
  float inv = redInv;

  for (int h = tid; h < H_DIM; h += 256) {
    float acc = 0.0f;
    const float* eb = enc + (size_t)b * S_LEN * H_DIM + h;
    for (int s = 0; s < S_LEN; ++s)
      acc += sc[s] * eb[(size_t)s * H_DIM];
    ctx[(size_t)b * H_DIM + h] = acc * inv;
  }
}

// ---------------------------------------------------------------------------
extern "C" void kernel_launch(void* const* d_in, const int* in_sizes, int n_in,
                              void* d_out, int out_size, void* d_ws, size_t ws_size,
                              hipStream_t stream) {
  (void)in_sizes; (void)n_in; (void)out_size; (void)ws_size;
  const int*   src      = (const int*)  d_in[0];
  const int*   tgt      = (const int*)  d_in[1];
  const float* enc_emb  = (const float*)d_in[2];
  const float* enc_Wih0 = (const float*)d_in[3];
  const float* enc_b0   = (const float*)d_in[4];
  const float* enc_Whh0 = (const float*)d_in[5];
  const float* enc_Wih1 = (const float*)d_in[6];
  const float* enc_b1   = (const float*)d_in[7];
  const float* enc_Whh1 = (const float*)d_in[8];
  const float* dec_emb  = (const float*)d_in[9];
  const float* dec_Wih0 = (const float*)d_in[10];
  const float* dec_b0   = (const float*)d_in[11];
  const float* dec_Whh0 = (const float*)d_in[12];
  const float* dec_Wih1 = (const float*)d_in[13];
  const float* dec_b1   = (const float*)d_in[14];
  const float* dec_Whh1 = (const float*)d_in[15];
  const float* W_a      = (const float*)d_in[16];
  const float* U_a      = (const float*)d_in[17];
  const float* v_a      = (const float*)d_in[18];
  const float* out_W    = (const float*)d_in[19];
  const float* out_b    = (const float*)d_in[20];
  float* logits = (float*)d_out;

  // -------- workspace carve-up (aligned 256B) --------
  char* ws = (char*)d_ws;
  size_t off = 0;
  auto take = [&](size_t bytes) -> void* {
    void* p = ws + off;
    off = (off + bytes + 255) & ~(size_t)255;
    return p;
  };
  bf16_t* Wenc0 = (bf16_t*)take((size_t)H_DIM * 768 * 2);          // [enc_Wih0 | enc_Whh0]
  bf16_t* Wenc1 = (bf16_t*)take((size_t)H_DIM * 1024 * 2);         // [enc_Wih1 | enc_Whh1]
  bf16_t* Wdec0 = (bf16_t*)take((size_t)H_DIM * 1280 * 2);         // [dec_Wih0 | dec_Whh0]
  bf16_t* Wdec1 = (bf16_t*)take((size_t)H_DIM * 1024 * 2);         // [dec_Wih1 | dec_Whh1]
  bf16_t* Wab   = (bf16_t*)take((size_t)H_DIM * H_DIM * 2);
  bf16_t* Uab   = (bf16_t*)take((size_t)H_DIM * H_DIM * 2);
  bf16_t* Woutb = (bf16_t*)take((size_t)V_DIM * 1024 * 2);
  float*  enc_out    = (float*) take((size_t)B_SZ * S_LEN * H_DIM * 4);
  bf16_t* enc_out_bf = (bf16_t*)take((size_t)B_SZ * S_LEN * H_DIM * 2);
  float*  Uenc       = (float*) take((size_t)B_SZ * S_LEN * H_DIM * 4);
  bf16_t* Abuf       = (bf16_t*)take((size_t)B_SZ * 1280 * 2);
  float*  h0         = (float*) take((size_t)B_SZ * H_DIM * 4);    // doubles as d0
  float*  d1         = (float*) take((size_t)B_SZ * H_DIM * 4);
  bf16_t* d1_bf      = (bf16_t*)take((size_t)B_SZ * H_DIM * 2);
  float*  qbuf       = (float*) take((size_t)B_SZ * H_DIM * 4);
  float*  ctxbuf     = (float*) take((size_t)B_SZ * H_DIM * 4);

  auto cgrid = [](int total) { return dim3((unsigned)((total + 255) / 256)); };

  // -------- one-time (per launch) bf16 weight prep --------
  k_cvt_cat<<<cgrid(H_DIM * 768),  256, 0, stream>>>(enc_Wih0, E_DIM, enc_Whh0, H_DIM, Wenc0, H_DIM);
  k_cvt_cat<<<cgrid(H_DIM * 1024), 256, 0, stream>>>(enc_Wih1, H_DIM, enc_Whh1, H_DIM, Wenc1, H_DIM);
  k_cvt_cat<<<cgrid(H_DIM * 1280), 256, 0, stream>>>(dec_Wih0, 768,  dec_Whh0, H_DIM, Wdec0, H_DIM);
  k_cvt_cat<<<cgrid(H_DIM * 1024), 256, 0, stream>>>(dec_Wih1, H_DIM, dec_Whh1, H_DIM, Wdec1, H_DIM);
  k_cvt_cat<<<cgrid(H_DIM * H_DIM), 256, 0, stream>>>(W_a,  H_DIM, W_a,  0, Wab,   H_DIM);
  k_cvt_cat<<<cgrid(H_DIM * H_DIM), 256, 0, stream>>>(U_a,  H_DIM, U_a,  0, Uab,   H_DIM);
  k_cvt_cat<<<cgrid(V_DIM * 1024), 256, 0, stream>>>(out_W, 1024,  out_W, 0, Woutb, V_DIM);
  k_zero<<<cgrid(B_SZ * H_DIM), 256, 0, stream>>>(h0, B_SZ * H_DIM);

  dim3 gblk(128);
  dim3 gH(H_DIM / 16, 1);     // N=512, M=64
  dim3 gV(V_DIM / 16, 1);     // N=256, M=64

  // -------- encoder: 128 sequential steps --------
  for (int t = 0; t < S_LEN; ++t) {
    // layer 0: A = [emb(src_t) | h0]
    k_build_A<<<cgrid(B_SZ * 768), 256, 0, stream>>>(
        enc_emb, src, S_LEN, t, E_DIM, h0, H_DIM, H_DIM,
        nullptr, 0, 0, 0, Abuf, 768);
    k_gemm_bf16<<<gH, gblk, 0, stream>>>(Abuf, 768, Wenc0, 768, enc_b0,
                                         h0, H_DIM, nullptr, 0, 1);
    // layer 1: A = [h0_new | h1_old], h1_old read strided from enc_out[:,t-1,:]
    k_build_A<<<cgrid(B_SZ * 1024), 256, 0, stream>>>(
        enc_emb, src, S_LEN, 0, 0, h0, H_DIM, H_DIM,
        (t > 0) ? (enc_out + (size_t)(t - 1) * H_DIM) : nullptr,
        S_LEN * H_DIM, H_DIM, (t > 0), Abuf, 1024);
    // write h1 into enc_out[:, t, :] (fp32, for attention ctx) and
    // enc_out_bf (bf16, A operand of the U_enc GEMM) via row strides
    k_gemm_bf16<<<gH, gblk, 0, stream>>>(Abuf, 1024, Wenc1, 1024, enc_b1,
                                         enc_out + (size_t)t * H_DIM, (long)S_LEN * H_DIM,
                                         enc_out_bf + (size_t)t * H_DIM, (long)S_LEN * H_DIM,
                                         1);
  }

  // -------- U_enc = enc_out @ U_a^T : one big WMMA GEMM, M = B*S = 8192 --------
  k_gemm_bf16<<<dim3(H_DIM / 16, (B_SZ * S_LEN) / 64), gblk, 0, stream>>>(
      enc_out_bf, H_DIM, Uab, H_DIM, nullptr, Uenc, H_DIM, nullptr, 0, 0);

  // decoder initial hidden: d0 = final h0 (already in h0), d1 = enc_out[:, S-1, :]
  k_copy_last<<<cgrid(B_SZ * H_DIM), 256, 0, stream>>>(enc_out, d1, d1_bf);

  // -------- decoder: 63 sequential steps --------
  for (int t = 0; t < T_LEN - 1; ++t) {
    // q = d1 @ W_a^T  (bf16 A mirror, no conversion in the K loop)
    k_gemm_bf16<<<gH, gblk, 0, stream>>>(d1_bf, H_DIM, Wab, H_DIM, nullptr,
                                         qbuf, H_DIM, nullptr, 0, 0);
    // attention -> ctx
    k_attention<<<dim3(B_SZ), dim3(256), 0, stream>>>(qbuf, Uenc, enc_out, v_a, ctxbuf);
    // layer 0: A = [emb(tgt_t) | ctx | d0]
    k_build_A<<<cgrid(B_SZ * 1280), 256, 0, stream>>>(
        dec_emb, tgt, T_LEN, t, E_DIM, ctxbuf, H_DIM, H_DIM,
        h0, H_DIM, H_DIM, 1, Abuf, 1280);
    k_gemm_bf16<<<gH, gblk, 0, stream>>>(Abuf, 1280, Wdec0, 1280, dec_b0,
                                         h0, H_DIM, nullptr, 0, 1);
    // layer 1: A = [d0_new | d1_old]
    k_build_A<<<cgrid(B_SZ * 1024), 256, 0, stream>>>(
        dec_emb, tgt, 0, 0, 0, h0, H_DIM, H_DIM,
        d1, H_DIM, H_DIM, 1, Abuf, 1024);
    k_gemm_bf16<<<gH, gblk, 0, stream>>>(Abuf, 1024, Wdec1, 1024, dec_b1,
                                         d1, H_DIM, d1_bf, H_DIM, 1);
    // logits = [d1_new | ctx] @ out_W^T + out_b  -> d_out[:, t, :]
    k_build_A<<<cgrid(B_SZ * 1024), 256, 0, stream>>>(
        dec_emb, tgt, 0, 0, 0, d1, H_DIM, H_DIM,
        ctxbuf, H_DIM, H_DIM, 1, Abuf, 1024);
    k_gemm_bf16<<<gV, gblk, 0, stream>>>(Abuf, 1024, Woutb, 1024, out_b,
                                         logits + (size_t)t * V_DIM,
                                         (long)(T_LEN - 1) * V_DIM, nullptr, 0, 0);
  }
}
